// ConcatCritic_62783831933329
// MI455X (gfx1250) — matrix operand order repaired and checked by reading, not compiled, not run on hardware
//
#include <hip/hip_runtime.h>
#include <hip/hip_bf16.h>

typedef __attribute__((ext_vector_type(16))) _Float16 v16h;
typedef __attribute__((ext_vector_type(8)))  float    v8f;
typedef __attribute__((ext_vector_type(4)))  _Float16 h4v;
typedef __attribute__((ext_vector_type(4)))  float    f4v;
typedef __attribute__((ext_vector_type(4)))  unsigned int u4v;
typedef __attribute__((ext_vector_type(4)))  unsigned int u32x4;
typedef __attribute__((ext_vector_type(8)))  int      i32x8;
typedef __attribute__((ext_vector_type(4)))  int      i32x4;

#define HH    512
#define NXW   128
#define MTOT  (512*512)
#define MT    128          // m-rows per workgroup
#define LDA   520          // A LDS stride (halves): 1040B, 16B-aligned, bank-spread
#define GC    128          // g-chunk width
#define KQ    128          // K staged per TDM transfer
#define LDB   136          // B LDS stride (halves) = KQ + 8 pad (TDM pad: 4 DW per 64 DW)

// ---------------- Layer 1: hx = x@W1x^T ; hyb = y@W1y^T + b1 ----------------
__global__ __launch_bounds__(256) void cc_prep_h1(
    const float* __restrict__ x, const float* __restrict__ y,
    const float* __restrict__ W1, const float* __restrict__ b1,
    float* __restrict__ hx, float* __restrict__ hyb)
{
    int t   = blockIdx.x * 256 + threadIdx.x;   // 0 .. 2*MTOT-1
    int sel = t >> 18;                          // 0: hx, 1: hyb
    int idx = t & (MTOT - 1);
    int r   = idx >> 9;
    int h   = idx & 511;
    const float* w = W1 + h * (2 * NXW) + sel * NXW;
    const float* v = (sel ? y : x) + r * NXW;
    float acc = 0.f;
#pragma unroll 8
    for (int n = 0; n < NXW; n += 4) {
        f4v wv = *(const f4v*)(w + n);
        f4v vv = *(const f4v*)(v + n);
        acc = fmaf(wv.x, vv.x, acc);
        acc = fmaf(wv.y, vv.y, acc);
        acc = fmaf(wv.z, vv.z, acc);
        acc = fmaf(wv.w, vv.w, acc);
    }
    if (sel) hyb[idx] = acc + b1[h];
    else     hx[idx]  = acc;
}

// ---------------- Convert W2 -> f16 (row-major [g][h]); write tail 0 --------
__global__ __launch_bounds__(256) void cc_prep_w2(
    const float* __restrict__ W2, _Float16* __restrict__ w2h,
    float* __restrict__ out, int out_size)
{
    int t = blockIdx.x * 256 + threadIdx.x;
    if (t < HH * HH) w2h[t] = (_Float16)W2[t];
    if (t == 0 && out_size > MTOT) out[MTOT] = 0.0f;   // second tuple output (0.0)
}

// Issue one TDM DMA: 128x128 f16 sub-tile of w2h -> LDS (padded rows, LDB=136h).
// D# group1 is constant: data_size=2B, pad_enable, pad_interval=64DW(code5),
// pad_amount=4DW(code3), tensor_dim0/1 = huge (no clip), tile 128x128, stride 512.
__device__ __forceinline__ void tdm_issue_b(const _Float16* gsrc, unsigned int lds_off)
{
    unsigned long long ga = (unsigned long long)(uintptr_t)gsrc;
    u32x4 g0;
    g0.x = 1u;                                               // count=1, user desc
    g0.y = lds_off;                                          // LDS byte offset
    g0.z = (unsigned int)ga;                                 // global_addr[31:0]
    g0.w = (unsigned int)((ga >> 32) & 0x1FFFFFFu)           // global_addr[56:32]
         | 0x80000000u;                                      // type=2 (bits 127:126)
    i32x8 g1 = {
        (int)0x07510000u,   // mask=0 | data_size=1(2B)<<16 | pad_en<<20 | intv5<<22 | amt3<<25
        (int)0xFFFF0000u,   // abar=0 | tensor_dim0[15:0]=0xFFFF <<16
        (int)0xFFFF7FFFu,   // tensor_dim0[31:16]=0x7FFF | tensor_dim1[15:0]<<16
        (int)0x00807FFFu,   // tensor_dim1[31:16]=0x7FFF | tile_dim0=128 <<16
        (int)128,           // tile_dim1=128 | tile_dim2=0
        (int)512,           // tensor_dim0_stride[31:0] = 512 elements
        0, 0                // stride hi / tensor_dim1_stride = 0
    };
    i32x4 gz = { 0, 0, 0, 0 };
#if __has_include(<hip/amd_detail/amd_gfx1250_TDM.h>)
    i32x8 gz8 = { 0, 0, 0, 0, 0, 0, 0, 0 };
    __builtin_amdgcn_tensor_load_to_lds(g0, g1, gz, gz, gz8, 0);
#else
    __builtin_amdgcn_tensor_load_to_lds(g0, g1, gz, gz, 0);
#endif
}

// ---------------- Fused main: h1 -> (WMMA GEMM vs W2) -> relu -> dot W3 -----
__global__ __launch_bounds__(256) void cc_main(
    const float* __restrict__ hx, const float* __restrict__ hyb,
    const _Float16* __restrict__ w2h,
    const float* __restrict__ b2, const float* __restrict__ W3,
    const float* __restrict__ b3, float* __restrict__ out)
{
    extern __shared__ char smem[];
    _Float16* Ah   = (_Float16*)smem;                 // [MT][LDA] h1 tile (f16)
    _Float16* Bq0  = Ah + MT * LDA;                   // [GC][LDB] W2 K-quarter, buf 0
    _Float16* Bq1  = Bq0 + GC * LDB;                  // buf 1
    float*    sred = (float*)(Bq1 + GC * LDB);        // [MT] cross-wave partials

    const int tid = threadIdx.x;
    const int m0  = blockIdx.x * MT;
    const int i   = m0 >> 9;
    const int j0  = m0 & 511;
    const int wv  = tid >> 5;

    const unsigned int bqoff[2] = {
        (unsigned int)(uintptr_t)(void*)Bq0,
        (unsigned int)(uintptr_t)(void*)Bq1
    };

    if (tid < MT) sred[tid] = 0.f;

    // Kick off first TDM transfer (chunk 0 -> buf 0); it overlaps A staging.
    if (wv == 0) tdm_issue_b(w2h, bqoff[0]);

    // Stage A: h1[row][h] = relu(hyb[i][h] + hx[j0+row][h]) -> f16
    for (int idx = tid; idx < MT * (HH / 4); idx += 256) {
        int row = idx >> 7;
        int h   = (idx & 127) * 4;
        f4v xv = *(const f4v*)(hx  + (size_t)(j0 + row) * HH + h);
        f4v yv = *(const f4v*)(hyb + (size_t)i * HH + h);
        h4v o;
        o.x = (_Float16)fmaxf(xv.x + yv.x, 0.f);
        o.y = (_Float16)fmaxf(xv.y + yv.y, 0.f);
        o.z = (_Float16)fmaxf(xv.z + yv.z, 0.f);
        o.w = (_Float16)fmaxf(xv.w + yv.w, 0.f);
        *(h4v*)(Ah + row * LDA + h) = o;
    }

    const int lane  = tid & 31;
    const int mrow0 = (wv & 3) * 32;             // 2 row-tiles per wave
    const int gwv   = (wv >> 2) * 64;            // 4 col-tiles per wave
    const int ln    = lane & 15;
    const int lg    = lane >> 4;

    float spart[2][8];
#pragma unroll
    for (int t = 0; t < 2; ++t)
#pragma unroll
        for (int v = 0; v < 8; ++v) spart[t][v] = 0.f;

    v8f acc[2][4];
    const v8f vzero = {};

    int chunk = 0;                               // chunk = gc*4 + kq
    for (int gc = 0; gc < 4; ++gc) {
#pragma unroll
        for (int t = 0; t < 2; ++t)
#pragma unroll
            for (int c = 0; c < 4; ++c) acc[t][c] = vzero;

        for (int kq = 0; kq < 4; ++kq, ++chunk) {
            __syncthreads();   // all waves done reading buf[(chunk+1)&1] (prev use)
            if (wv == 0) {
                if (chunk + 1 < 16) {
                    int nc = chunk + 1;
                    tdm_issue_b(w2h + (size_t)((nc >> 2) * GC) * HH + (nc & 3) * KQ,
                                bqoff[nc & 1]);
                    __builtin_amdgcn_s_wait_tensorcnt(1);  // chunk's DMA done
                } else {
                    __builtin_amdgcn_s_wait_tensorcnt(0);
                }
            }
            __syncthreads();   // broadcast: buf[chunk&1] is ready
            const _Float16* Bq = (chunk & 1) ? Bq1 : Bq0;

#pragma unroll
            for (int k0q = 0; k0q < KQ; k0q += 32) {
                v16h a[2], b[4];
#pragma unroll
                for (int t = 0; t < 2; ++t) {
                    // A 16x32 f16: lane m=ln, halves 0..7 = K kg..kg+7,
                    // halves 8..15 = K kg+16..kg+23, kg = lg*8
                    const _Float16* p = Ah + (mrow0 + t * 16 + ln) * LDA
                                      + kq * KQ + k0q + lg * 8;
                    union { v16h v; struct { u4v lo, hi; } q; } u;
                    u.q.lo = *(const u4v*)p;
                    u.q.hi = *(const u4v*)(p + 16);
                    a[t] = u.v;
                }
#pragma unroll
                for (int c = 0; c < 4; ++c) {
                    // B 32x16 f16: lane col n=ln, halves 0..7 = K kst..kst+7,
                    // halves 8..15 = K kst+16..kst+23, kst = k0q + lg*8
                    const _Float16* p = Bq + (gwv + c * 16 + ln) * LDB
                                      + k0q + lg * 8;
                    union { v16h v; struct { u4v lo, hi; } q; } u;
                    u.q.lo = *(const u4v*)p;
                    u.q.hi = *(const u4v*)(p + 16);
                    b[c] = u.v;
                }
#pragma unroll
                for (int t = 0; t < 2; ++t)
#pragma unroll
                    for (int c = 0; c < 4; ++c)
                        acc[t][c] = __builtin_amdgcn_wmma_f32_16x16x32_f16(
                            false, a[t], false, b[c], (short)0, acc[t][c],
                            false, false);
            }
        }

        // Epilogue for this g-chunk: +b2, relu, scale by W3[g], fold into partials
#pragma unroll
        for (int t = 0; t < 2; ++t)
#pragma unroll
            for (int c = 0; c < 4; ++c) {
                int g = gc * GC + gwv + c * 16 + ln;
                float bbv = b2[g];
                float wwv = W3[g];
#pragma unroll
                for (int v = 0; v < 8; ++v) {
                    float h2v = fmaxf(acc[t][c][v] + bbv, 0.f);
                    spart[t][v] = fmaf(h2v, wwv, spart[t][v]);
                }
            }
    }

    // Column reduction within each 16-lane group, then merge the two g-half
    // waves via LDS (exactly 2 commutative f32 adds per slot -> deterministic).
#pragma unroll
    for (int t = 0; t < 2; ++t)
#pragma unroll
        for (int v = 0; v < 8; ++v) {
            float r = spart[t][v];
            r += __shfl_xor(r, 1, 32);
            r += __shfl_xor(r, 2, 32);
            r += __shfl_xor(r, 4, 32);
            r += __shfl_xor(r, 8, 32);
            if (ln == 0) atomicAdd(&sred[mrow0 + t * 16 + lg * 8 + v], r);
        }
    __syncthreads();

    if (tid < MT) out[m0 + tid] = sred[tid] + b3[0];
}

extern "C" void kernel_launch(void* const* d_in, const int* in_sizes, int n_in,
                              void* d_out, int out_size, void* d_ws, size_t ws_size,
                              hipStream_t stream) {
    const float* x  = (const float*)d_in[0];
    const float* y  = (const float*)d_in[1];
    const float* W1 = (const float*)d_in[2];
    const float* b1 = (const float*)d_in[3];
    const float* W2 = (const float*)d_in[4];
    const float* b2 = (const float*)d_in[5];
    const float* W3 = (const float*)d_in[6];
    const float* b3 = (const float*)d_in[7];
    float* out = (float*)d_out;

    float*    hx  = (float*)d_ws;               // [512*512] f32
    float*    hyb = hx + MTOT;                  // [512*512] f32
    _Float16* w2h = (_Float16*)(hyb + MTOT);    // [512*512] f16

    cc_prep_h1<<<(2 * MTOT) / 256, 256, 0, stream>>>(x, y, W1, b1, hx, hyb);
    cc_prep_w2<<<MTOT / 256, 256, 0, stream>>>(W2, w2h, out, out_size);

    size_t lds = (size_t)(MT * LDA + 2 * GC * LDB) * sizeof(_Float16)
               + MT * sizeof(float);
    cc_main<<<MTOT / MT, 256, lds, stream>>>(hx, hyb, w2h, b2, W3, b3, out);
}
